// generative_model_25116968747266
// MI455X (gfx1250) — matrix operand order repaired and checked by reading, not compiled, not run on hardware
//
#include <hip/hip_runtime.h>
#include <hip/hip_bf16.h>

#define G_  128
#define N_  512
#define ZI_ 128
#define DD_ 64

typedef __attribute__((ext_vector_type(16))) _Float16 v16h;
typedef __attribute__((ext_vector_type(8)))  _Float16 v8h;
typedef __attribute__((ext_vector_type(8)))  float    v8f;
typedef __attribute__((ext_vector_type(4)))  float    v4f;
typedef __attribute__((ext_vector_type(4)))  unsigned v4u;

// ---- WMMA helper: D = A(16x32 f16) * B(32x16 f16) + C(16x16 f32) ----
static __device__ __forceinline__ v8f wmma_f16(v16h a, v16h b, v8f c) {
  return __builtin_amdgcn_wmma_f32_16x16x32_f16(
      /*neg_a=*/false, a, /*neg_b=*/false, b,
      /*c_mod=*/(short)0, c, /*reuse_a=*/false, /*reuse_b=*/false);
}

// A-fragment (16x32, MxK): lane holds row M=lane%16.
// lanes 0-15 hold K={c0..c0+15} with c0=0, lanes 16-31 c0=16.
static __device__ __forceinline__ v16h load_a_frag(const _Float16* rowp, int lane) {
  const int c0 = (lane & 16) ? 16 : 0;
  v8h lo = *(const v8h*)(rowp + c0);
  v8h hi = *(const v8h*)(rowp + c0 + 8);
  v16h a;
#pragma unroll
  for (int j = 0; j < 8; ++j) { a[j] = lo[j]; a[j + 8] = hi[j]; }
  return a;
}

// B-fragment where B = Mrows^T: B[k][n] = mat[nbase+n][kbase+k], row-major stride 64.
static __device__ __forceinline__ v16h load_bT_frag(const _Float16* mat, int nbase,
                                                    int kbase, int lane) {
  const int n = nbase + (lane & 15);
  const int k = kbase + ((lane & 16) ? 16 : 0);
  return *(const v16h*)(mat + (size_t)n * DD_ + k);
}

#define LDS_FENCE() asm volatile("s_wait_dscnt 0" ::: "memory")

// ================= Kernel 1: Q/K/V = Z*W + b (Q,K row-major f16; V transposed) ==========
__global__ __launch_bounds__(256) void qkv_proj_kernel(
    const float* __restrict__ z,
    const float* __restrict__ Wq, const float* __restrict__ bq,
    const float* __restrict__ Wk, const float* __restrict__ bk,
    const float* __restrict__ Wv, const float* __restrict__ bv,
    _Float16* __restrict__ Qh, _Float16* __restrict__ Kh, _Float16* __restrict__ Vt) {
  __shared__ _Float16 Wt[DD_][ZI_];       // one weight matrix (transposed), 16 KB
  __shared__ _Float16 Stg[8][16 * DD_];   // per-wave 16x64 output staging, 16 KB

  const int tid = threadIdx.x, wave = tid >> 5, lane = tid & 31;
  const int g = blockIdx.y;
  const int row0 = blockIdx.x * 128 + wave * 16;
  const int colc = lane & 15, hi8 = (lane & 16) ? 8 : 0;

  const float* Ws[3] = {Wq, Wk, Wv};
  const float* Bs[3] = {bq, bk, bv};

  // A fragments of Z (f32 -> f16), 4 k-chunks of 32
  const float* zr = z + ((size_t)g * N_ + row0 + colc) * ZI_;
  v16h aZ[4];
#pragma unroll
  for (int kc = 0; kc < 4; ++kc) {
    const int c0 = (lane & 16) ? 16 : 0;
    const float* p0 = zr + kc * 32 + c0;
    const float* p1 = p0 + 8;
#pragma unroll
    for (int j = 0; j < 8; ++j) { aZ[kc][j] = (_Float16)p0[j]; aZ[kc][j + 8] = (_Float16)p1[j]; }
  }

  for (int m = 0; m < 3; ++m) {
    __syncthreads();  // protect Wt (and per-wave Stg) from previous iteration
    for (int i = tid; i < ZI_ * DD_; i += 256) {
      int k = i >> 6, n = i & 63;  // W is [128][64] row-major
      Wt[n][k] = (_Float16)Ws[m][i];
    }
    __syncthreads();

#pragma unroll
    for (int nt = 0; nt < 4; ++nt) {
      v16h bw[4];
#pragma unroll
      for (int kc = 0; kc < 4; ++kc)
        bw[kc] = *(const v16h*)&Wt[nt * 16 + colc][kc * 32 + ((lane & 16) ? 16 : 0)];
      v8f c = {0.f, 0.f, 0.f, 0.f, 0.f, 0.f, 0.f, 0.f};
#pragma unroll
      for (int kc = 0; kc < 4; ++kc) c = wmma_f16(aZ[kc], bw[kc], c);

      const float bb = Bs[m][nt * 16 + colc];
      if (m < 2) {  // stage row-major [16][64]
#pragma unroll
        for (int r = 0; r < 8; ++r)
          Stg[wave][(r + hi8) * DD_ + nt * 16 + colc] = (_Float16)(c[r] + bb);
      } else {      // stage transposed [64][16] for Vt
#pragma unroll
        for (int r = 0; r < 8; ++r)
          Stg[wave][(nt * 16 + colc) * 16 + (r + hi8)] = (_Float16)(c[r] + bb);
      }
    }
    LDS_FENCE();

    if (m < 2) {
      // whole 16-row strip is 2 KB contiguous: 4x b128 per lane
      _Float16* dst = (m == 0 ? Qh : Kh) + ((size_t)g * N_ + row0) * DD_;
      const v4u* src = (const v4u*)&Stg[wave][0];
      v4u* d4 = (v4u*)dst;
#pragma unroll
      for (int j = 0; j < 4; ++j) d4[lane * 4 + j] = src[lane * 4 + j];
    } else {
      // Vt: each d-row is a 32B chunk at stride 1 KB
#pragma unroll
      for (int j = 0; j < 2; ++j) {
        const int d = lane + j * 32;
        const v4u* src = (const v4u*)&Stg[wave][d * 16];
        v4u* dv = (v4u*)(Vt + ((size_t)g * DD_ + d) * N_ + row0);
        dv[0] = src[0];
        dv[1] = src[1];
      }
    }
  }
}

// ============== Kernel 2: Z_att = softmax(Q K^T) V / 8  (flash-style online softmax) ====
__global__ __launch_bounds__(256) void attn_kernel(
    const _Float16* __restrict__ Qh, const _Float16* __restrict__ Kh,
    const _Float16* __restrict__ Vt, _Float16* __restrict__ Za) {
  __shared__ _Float16 P[8][16][40];       // per-wave 16x32 P tile, padded stride
  __shared__ _Float16 Stg[8][16 * DD_];   // per-wave Z_att staging for coalesced store

  const int tid = threadIdx.x, wave = tid >> 5, lane = tid & 31;
  const int g = blockIdx.y;
  const int row0 = blockIdx.x * 128 + wave * 16;
  const int colc = lane & 15, hi8 = (lane & 16) ? 8 : 0;

  const _Float16* Qg = Qh + (size_t)g * N_ * DD_;
  const _Float16* Kg = Kh + (size_t)g * N_ * DD_;
  const _Float16* Vg = Vt + (size_t)g * DD_ * N_;

  const _Float16* qrow = Qg + (size_t)(row0 + colc) * DD_;
  const v16h aq0 = load_a_frag(qrow, lane);
  const v16h aq1 = load_a_frag(qrow + 32, lane);

  float mrow[8], lrow[8];
  const v8f vz = {0.f, 0.f, 0.f, 0.f, 0.f, 0.f, 0.f, 0.f};
  v8f o[4] = {vz, vz, vz, vz};
#pragma unroll
  for (int r = 0; r < 8; ++r) { mrow[r] = -1e30f; lrow[r] = 0.f; }

  for (int kb = 0; kb < N_; kb += 32) {
    // load all K fragments first, then chain WMMAs (one wait covers all loads)
    const v16h b00 = load_bT_frag(Kg, kb, 0, lane);
    const v16h b01 = load_bT_frag(Kg, kb, 32, lane);
    const v16h b10 = load_bT_frag(Kg, kb + 16, 0, lane);
    const v16h b11 = load_bT_frag(Kg, kb + 16, 32, lane);
    v8f s0 = vz, s1 = vz;
    s0 = wmma_f16(aq0, b00, s0);
    s0 = wmma_f16(aq1, b01, s0);
    s1 = wmma_f16(aq0, b10, s1);
    s1 = wmma_f16(aq1, b11, s1);

#pragma unroll
    for (int r = 0; r < 8; ++r) {
      float mb = fmaxf(s0[r], s1[r]);
      mb = fmaxf(mb, __shfl_xor(mb, 1, 32));
      mb = fmaxf(mb, __shfl_xor(mb, 2, 32));
      mb = fmaxf(mb, __shfl_xor(mb, 4, 32));
      mb = fmaxf(mb, __shfl_xor(mb, 8, 32));
      const float mn = fmaxf(mrow[r], mb);
      const float sc = expf(mrow[r] - mn);
      const float e0 = expf(s0[r] - mn);
      const float e1 = expf(s1[r] - mn);
      float rs = e0 + e1;
      rs += __shfl_xor(rs, 1, 32);
      rs += __shfl_xor(rs, 2, 32);
      rs += __shfl_xor(rs, 4, 32);
      rs += __shfl_xor(rs, 8, 32);
      mrow[r] = mn;
      lrow[r] = lrow[r] * sc + rs;
      o[0][r] *= sc; o[1][r] *= sc; o[2][r] *= sc; o[3][r] *= sc;
      P[wave][r + hi8][colc]      = (_Float16)e0;   // D-layout -> row-major in LDS
      P[wave][r + hi8][16 + colc] = (_Float16)e1;
    }
    LDS_FENCE();
    const v16h ap = load_a_frag(&P[wave][colc][0], lane);  // re-read in A-layout
    v16h bv[4];
#pragma unroll
    for (int t = 0; t < 4; ++t)
      bv[t] = *(const v16h*)(Vg + (size_t)(t * 16 + colc) * N_ + kb +
                             ((lane & 16) ? 16 : 0));
#pragma unroll
    for (int t = 0; t < 4; ++t) o[t] = wmma_f16(ap, bv[t], o[t]);
    LDS_FENCE();
  }

  // stage result, then 4x b128 coalesced store of the contiguous 2 KB strip
#pragma unroll
  for (int r = 0; r < 8; ++r) {
    const float inv = 1.0f / (lrow[r] * 8.0f);  // softmax norm and /sqrt(64)
#pragma unroll
    for (int t = 0; t < 4; ++t)
      Stg[wave][(r + hi8) * DD_ + t * 16 + colc] = (_Float16)(o[t][r] * inv);
  }
  LDS_FENCE();
  _Float16* dst = Za + ((size_t)g * N_ + row0) * DD_;
  const v4u* src = (const v4u*)&Stg[wave][0];
  v4u* d4 = (v4u*)dst;
#pragma unroll
  for (int j = 0; j < 4; ++j) d4[lane * 4 + j] = src[lane * 4 + j];
}

// ============== Kernel 3: A_hat = sigmoid(Za Za^T), zero diagonal (128 MB stream-out) ===
__global__ __launch_bounds__(256) void ahat_kernel(const _Float16* __restrict__ Za,
                                                   float* __restrict__ out) {
  const int tid = threadIdx.x, wave = tid >> 5, lane = tid & 31;
  const int g = blockIdx.y;
  const int row0 = blockIdx.x * 128 + wave * 16;
  const int colc = lane & 15, hi8 = (lane & 16) ? 8 : 0;

  const _Float16* Zg = Za + (size_t)g * N_ * DD_;
  const _Float16* arow = Zg + (size_t)(row0 + colc) * DD_;
  const v16h a0 = load_a_frag(arow, lane);
  const v16h a1 = load_a_frag(arow + 32, lane);
  float* outA = out + (size_t)g * N_ * N_;
  const v8f vz = {0.f, 0.f, 0.f, 0.f, 0.f, 0.f, 0.f, 0.f};

  for (int nb = 0; nb < N_; nb += 16) {
    const v16h b0 = load_bT_frag(Zg, nb, 0, lane);
    const v16h b1 = load_bT_frag(Zg, nb, 32, lane);
    v8f c = vz;
    c = wmma_f16(a0, b0, c);
    c = wmma_f16(a1, b1, c);
    const int ccol = nb + colc;
#pragma unroll
    for (int r = 0; r < 8; ++r) {
      const int rrow = row0 + r + hi8;
      float v = 1.0f / (1.0f + expf(-c[r]));
      if (rrow == ccol) v = 0.0f;
      // streaming 128 MB, never re-read on device: non-temporal to keep Za hot in L2
      __builtin_nontemporal_store(v, &outA[(size_t)rrow * N_ + ccol]);
    }
  }
}

// ============== Kernel 4: X_hat = 0 =====================================================
__global__ void zero_xhat_kernel(float* __restrict__ p, long long n4) {
  long long i = (long long)blockIdx.x * blockDim.x + threadIdx.x;
  const v4f z = {0.f, 0.f, 0.f, 0.f};
  if (i < n4) __builtin_nontemporal_store(z, (v4f*)p + i);
}

extern "C" void kernel_launch(void* const* d_in, const int* in_sizes, int n_in,
                              void* d_out, int out_size, void* d_ws, size_t ws_size,
                              hipStream_t stream) {
  const float* z  = (const float*)d_in[0];
  // d_in[1] = batch (int64): sorted equal-size graphs -> pure reshape, unused.
  const float* Wq = (const float*)d_in[2];
  const float* bq = (const float*)d_in[3];
  const float* Wk = (const float*)d_in[4];
  const float* bk = (const float*)d_in[5];
  const float* Wv = (const float*)d_in[6];
  const float* bv = (const float*)d_in[7];
  float* out = (float*)d_out;

  const size_t QE = (size_t)G_ * N_ * DD_;  // 4M halves each
  _Float16* Qh = (_Float16*)d_ws;
  _Float16* Kh = Qh + QE;
  _Float16* Vt = Kh + QE;
  _Float16* Za = Vt + QE;  // total workspace use: 32 MB

  dim3 grid(4, G_);  // 4 row-blocks of 128 x 128 graphs = 512 workgroups
  qkv_proj_kernel<<<grid, 256, 0, stream>>>(z, Wq, bq, Wk, bk, Wv, bv, Qh, Kh, Vt);
  attn_kernel<<<grid, 256, 0, stream>>>(Qh, Kh, Vt, Za);
  ahat_kernel<<<grid, 256, 0, stream>>>(Za, out);

  const long long ahat_elems = (long long)G_ * N_ * N_;        // 33,554,432
  const long long xhat_n4    = ((long long)G_ * N_ * ZI_) / 4; // 2,097,152 x float4
  zero_xhat_kernel<<<(unsigned)((xhat_n4 + 255) / 256), 256, 0, stream>>>(out + ahat_elems,
                                                                          xhat_n4);
}
